// GaussianSampling_21749714387345
// MI455X (gfx1250) — matrix operand order repaired and checked by reading, not compiled, not run on hardware
//
#include <hip/hip_runtime.h>
#include <math.h>

typedef __attribute__((ext_vector_type(16))) _Float16 v16h;
typedef __attribute__((ext_vector_type(8)))  float    v8f;

#define D_MODEL 512
#define HEADS   16
#define DHEAD   32
#define B_SZ    4
#define L1_LEN  1024
#define L_LEN   8192
#define NW      512
#define DFF     2048

union Frag {
    v16h h;
    unsigned int u[8];
};

// pack two f32 -> one dword of two f16 (v_cvt_pk_rtz_f16_f32)
static __device__ __forceinline__ unsigned int pk2(float a, float b) {
    union { __fp16 __attribute__((ext_vector_type(2))) h2; unsigned int u; } c;
    c.h2 = __builtin_amdgcn_cvt_pkrtz(a, b);
    return c.u;
}

// branch-free gelu (tanh approx) built on v_exp_f32
static __device__ __forceinline__ float gelu_tanh(float x) {
    float y = 0.7978845608028654f * (x + 0.044715f * x * x * x);
    float e = __expf(-2.0f * fabsf(y));
    float t = (1.0f - e) / (1.0f + e);
    t = copysignf(t, y);
    return 0.5f * x * (1.0f + t);
}

// ---------------------------------------------------------------------------
// statT[b, n, c] = |max - min| over window of 16 along L in global_feat[b, c, :]
// ---------------------------------------------------------------------------
__global__ __launch_bounds__(256) void winstat_kernel(const float* __restrict__ gf,
                                                      float* __restrict__ statT) {
    int gid = blockIdx.x * 256 + threadIdx.x;           // B*D*NW threads
    int b   = gid >> 18;
    int rem = gid & ((1 << 18) - 1);
    int c   = rem >> 9;
    int n   = rem & 511;
    const float* p = gf + ((size_t)(b * D_MODEL + c)) * L_LEN + n * 16;
    float mx = p[0], mn = p[0];
#pragma unroll
    for (int j = 1; j < 16; j++) { float t = p[j]; mx = fmaxf(mx, t); mn = fminf(mn, t); }
    statT[((size_t)(b * NW + n)) * D_MODEL + c] = fabsf(mx - mn);
}

// ---------------------------------------------------------------------------
// WMMA GEMM: C[M,N] = act(A[M,K] @ W[N,K]^T + bias[N])
// Block: 256 thr (8 waves), tile 128x64, K-step 32. Each wave -> 32x32 tile
// (2 A-frags x 2 B-frags -> 4 v_wmma per K-step, fragment reuse).
// Both LDS tiles stored K-contiguous with padded row stride 40 f16 (20 dwords:
// conflict-free, 16B-aligned rows) so fragments assemble as ds_load_b128.
// act: 0=none 1=relu 2=gelu
// ---------------------------------------------------------------------------
__global__ __launch_bounds__(256) void gemm_bias_act(const float* __restrict__ A,
                                                     const float* __restrict__ W,
                                                     const float* __restrict__ bias,
                                                     float* __restrict__ C,
                                                     int M, int N, int K, int act) {
    __shared__ _Float16 sA[128 * 40];   // [m][k]
    __shared__ _Float16 sBT[64 * 40];   // [n][k]

    const int tid  = threadIdx.x;
    const int m0   = blockIdx.x * 128;
    const int n0   = blockIdx.y * 64;
    const int lane = tid & 31;
    const int wid  = tid >> 5;
    const int h4   = lane >> 4;
    const int ml   = lane & 15;
    const int wm   = (wid & 3) * 32;    // 4x2 waves cover 128x64
    const int wn   = (wid >> 2) * 32;

    const int arow = tid >> 1;          // 0..127
    const int aoff = (tid & 1) * 16;    // 16 floats each
    const int brow = tid >> 2;          // 0..63
    const int boff = (tid & 3) * 8;     // 8 floats each

    v8f c00 = {}, c01 = {}, c10 = {}, c11 = {};

    for (int kk = 0; kk < K; kk += 32) {
        const float* ap = A + (size_t)(m0 + arow) * K + kk + aoff;
        unsigned int* da = (unsigned int*)&sA[arow * 40 + aoff];
#pragma unroll
        for (int j = 0; j < 8; j++) da[j] = pk2(ap[2 * j], ap[2 * j + 1]);

        const float* wp = W + (size_t)(n0 + brow) * K + kk + boff;
        unsigned int* db = (unsigned int*)&sBT[brow * 40 + boff];
#pragma unroll
        for (int j = 0; j < 4; j++) db[j] = pk2(wp[2 * j], wp[2 * j + 1]);

        // prefetch next K tile (global_prefetch_b8, speculative)
        __builtin_prefetch(ap + 32, 0, 1);
        __builtin_prefetch(wp + 32, 0, 1);

        __syncthreads();

        Frag a0, a1, b0, b1;
        const unsigned int* pa0 = (const unsigned int*)&sA[(wm + ml) * 40];
        const unsigned int* pa1 = (const unsigned int*)&sA[(wm + 16 + ml) * 40];
        const unsigned int* pb0 = (const unsigned int*)&sBT[(wn + ml) * 40];
        const unsigned int* pb1 = (const unsigned int*)&sBT[(wn + 16 + ml) * 40];
#pragma unroll
        for (int v = 0; v < 8; v++) {
            int ka = (2 * v + (h4 ? (v < 4 ? 8 : 16) : (v < 4 ? 0 : 8))) >> 1;
            int kb = (2 * v + (h4 ? (v < 4 ? 8 : 16) : (v < 4 ? 0 : 8))) >> 1;
            a0.u[v] = pa0[ka];
            a1.u[v] = pa1[ka];
            b0.u[v] = pb0[kb];
            b1.u[v] = pb1[kb];
        }
        c00 = __builtin_amdgcn_wmma_f32_16x16x32_f16(false, a0.h, false, b0.h, (short)0, c00, false, false);
        c01 = __builtin_amdgcn_wmma_f32_16x16x32_f16(false, a0.h, false, b1.h, (short)0, c01, false, false);
        c10 = __builtin_amdgcn_wmma_f32_16x16x32_f16(false, a1.h, false, b0.h, (short)0, c10, false, false);
        c11 = __builtin_amdgcn_wmma_f32_16x16x32_f16(false, a1.h, false, b1.h, (short)0, c11, false, false);
        __syncthreads();
    }

    const int ng0 = n0 + wn + ml;
    const int ng1 = ng0 + 16;
    const float bb0 = bias[ng0], bb1 = bias[ng1];
#pragma unroll
    for (int r = 0; r < 8; r++) {
        int mA = m0 + wm + r + 8 * h4;       // rows of c00/c01
        int mB = mA + 16;                    // rows of c10/c11
        float x00 = c00[r] + bb0, x01 = c01[r] + bb1;
        float x10 = c10[r] + bb0, x11 = c11[r] + bb1;
        if (act == 1) {
            x00 = fmaxf(x00, 0.f); x01 = fmaxf(x01, 0.f);
            x10 = fmaxf(x10, 0.f); x11 = fmaxf(x11, 0.f);
        } else if (act == 2) {
            x00 = gelu_tanh(x00); x01 = gelu_tanh(x01);
            x10 = gelu_tanh(x10); x11 = gelu_tanh(x11);
        }
        C[(size_t)mA * N + ng0] = x00;
        C[(size_t)mA * N + ng1] = x01;
        C[(size_t)mB * N + ng0] = x10;
        C[(size_t)mB * N + ng1] = x11;
    }
}

// ---------------------------------------------------------------------------
// Flash attention: one wave per 16-query tile, streams 32 keys per iter.
// ---------------------------------------------------------------------------
__global__ __launch_bounds__(256) void flash_attn_kernel(const float* __restrict__ Q,
                                                         const float* __restrict__ K,
                                                         const float* __restrict__ V,
                                                         float* __restrict__ O,
                                                         int Lq, int Lk) {
    __shared__ _Float16 sP[8 * 16 * 40];
    const int tid = threadIdx.x, lane = tid & 31, wid = tid >> 5;
    const int h4 = lane >> 4, ml = lane & 15;
    _Float16* lP = sP + wid * 16 * 40;

    const int gw     = blockIdx.x * 8 + wid;
    const int qtiles = Lq >> 4;
    const int qt = gw % qtiles;
    const int h  = (gw / qtiles) % HEADS;
    const int b  = gw / (qtiles * HEADS);
    const int hb = h * DHEAD;
    const float scale = 0.17677669529663687f;   // 1/sqrt(32)

    // Q fragment (A layout), held in registers for the whole K loop
    const float* qrow = Q + ((size_t)(b * Lq + qt * 16 + ml)) * D_MODEL + hb;
    Frag qf;
#pragma unroll
    for (int v = 0; v < 8; v++) {
        int k = 2 * v + (h4 ? (v < 4 ? 8 : 16) : (v < 4 ? 0 : 8));
        qf.u[v] = pk2(qrow[k], qrow[k + 1]);
    }

    float mi[8], li[8];
#pragma unroll
    for (int r = 0; r < 8; r++) { mi[r] = -1e30f; li[r] = 0.f; }
    v8f o0 = {}, o1 = {};

    for (int kt = 0; kt < Lk; kt += 32) {
        // K^T fragments: B[k=dim][n=key]; lane supplies key kt+ml
        const float* krow0 = K + ((size_t)(b * Lk + kt + ml)) * D_MODEL + hb;
        const float* krow1 = krow0 + (size_t)16 * D_MODEL;
        Frag kf0, kf1;
#pragma unroll
        for (int v = 0; v < 8; v++) {
            int kd = 2 * v + (h4 ? (v < 4 ? 8 : 16) : (v < 4 ? 0 : 8));
            kf0.u[v] = pk2(krow0[kd], krow0[kd + 1]);
            kf1.u[v] = pk2(krow1[kd], krow1[kd + 1]);
        }
        v8f z = {};
        v8f s0 = __builtin_amdgcn_wmma_f32_16x16x32_f16(false, qf.h, false, kf0.h, (short)0, z, false, false);
        v8f s1 = __builtin_amdgcn_wmma_f32_16x16x32_f16(false, qf.h, false, kf1.h, (short)0, z, false, false);

        float alpha[8];
#pragma unroll
        for (int r = 0; r < 8; r++) {   // row M = r + 8*h4 lives across 16 lanes of half
            float a0 = s0[r] * scale, a1 = s1[r] * scale;
            float rmx = fmaxf(a0, a1);
            rmx = fmaxf(rmx, __shfl_xor(rmx, 1, 32));
            rmx = fmaxf(rmx, __shfl_xor(rmx, 2, 32));
            rmx = fmaxf(rmx, __shfl_xor(rmx, 4, 32));
            rmx = fmaxf(rmx, __shfl_xor(rmx, 8, 32));
            float mn = fmaxf(mi[r], rmx);
            float p0 = __expf(a0 - mn), p1 = __expf(a1 - mn);
            float rs = p0 + p1;
            rs += __shfl_xor(rs, 1, 32);
            rs += __shfl_xor(rs, 2, 32);
            rs += __shfl_xor(rs, 4, 32);
            rs += __shfl_xor(rs, 8, 32);
            float al = __expf(mi[r] - mn);
            li[r] = li[r] * al + rs;
            mi[r] = mn;
            alpha[r] = al;
            // P row = query m, col = local key; stored f16, k-contiguous
            lP[(r + 8 * h4) * 40 + ml]      = (_Float16)p0;
            lP[(r + 8 * h4) * 40 + 16 + ml] = (_Float16)p1;
        }
        // cross-lane LDS RAW within the wave: wait for ds writes to land
        __asm__ volatile("s_wait_dscnt 0x0" ::: "memory");

#pragma unroll
        for (int r = 0; r < 8; r++) { o0[r] *= alpha[r]; o1[r] *= alpha[r]; }

        Frag pf;   // P as A-fragment: row m = lane%16, dword loads
        const unsigned int* pp = (const unsigned int*)&lP[ml * 40];
#pragma unroll
        for (int v = 0; v < 8; v++) {
            int ka = (2 * v + (h4 ? (v < 4 ? 8 : 16) : (v < 4 ? 0 : 8))) >> 1;
            pf.u[v] = pp[ka];
        }
        // V fragments: B[k=key][n=dim]
        const float* vbase = V + ((size_t)(b * Lk + kt)) * D_MODEL + hb;
        Frag vf0, vf1;
#pragma unroll
        for (int v = 0; v < 8; v++) {
            int kk = 2 * v + (h4 ? (v < 4 ? 8 : 16) : (v < 4 ? 0 : 8));
            vf0.u[v] = pk2(vbase[(size_t)kk * D_MODEL + ml],
                           vbase[(size_t)(kk + 1) * D_MODEL + ml]);
            vf1.u[v] = pk2(vbase[(size_t)kk * D_MODEL + 16 + ml],
                           vbase[(size_t)(kk + 1) * D_MODEL + 16 + ml]);
        }
        o0 = __builtin_amdgcn_wmma_f32_16x16x32_f16(false, pf.h, false, vf0.h, (short)0, o0, false, false);
        o1 = __builtin_amdgcn_wmma_f32_16x16x32_f16(false, pf.h, false, vf1.h, (short)0, o1, false, false);
    }

    float* obase = O + ((size_t)(b * Lq + qt * 16)) * D_MODEL + hb;
#pragma unroll
    for (int r = 0; r < 8; r++) {
        float inv = 1.f / li[r];
        obase[(size_t)(r + 8 * h4) * D_MODEL + ml]      = o0[r] * inv;
        obase[(size_t)(r + 8 * h4) * D_MODEL + 16 + ml] = o1[r] * inv;
    }
}

// ---------------------------------------------------------------------------
// out = LayerNorm(X + Y) * g + b ; one block per row of 512.
// transpose_out=1 writes [B, D, L1] (final output layout).
// ---------------------------------------------------------------------------
__global__ __launch_bounds__(256) void add_ln_kernel(const float* __restrict__ X,
                                                     const float* __restrict__ Y,
                                                     const float* __restrict__ g,
                                                     const float* __restrict__ bt,
                                                     float* __restrict__ out,
                                                     int transpose_out) {
    __shared__ float red[256];
    __shared__ float s_mu, s_rs;
    const int row = blockIdx.x, tid = threadIdx.x;
    const float* xr = X + (size_t)row * D_MODEL;
    const float* yr = Y + (size_t)row * D_MODEL;
    float a0 = xr[tid] + yr[tid];
    float a1 = xr[tid + 256] + yr[tid + 256];

    red[tid] = a0 + a1;
    __syncthreads();
    for (int s = 128; s > 0; s >>= 1) { if (tid < s) red[tid] += red[tid + s]; __syncthreads(); }
    if (tid == 0) s_mu = red[0] * (1.0f / D_MODEL);
    __syncthreads();
    float mu = s_mu;
    float d0 = a0 - mu, d1 = a1 - mu;
    red[tid] = d0 * d0 + d1 * d1;
    __syncthreads();
    for (int s = 128; s > 0; s >>= 1) { if (tid < s) red[tid] += red[tid + s]; __syncthreads(); }
    if (tid == 0) s_rs = rsqrtf(red[0] * (1.0f / D_MODEL) + 1e-5f);
    __syncthreads();
    float rs = s_rs;
    float v0 = d0 * rs * g[tid] + bt[tid];
    float v1 = d1 * rs * g[tid + 256] + bt[tid + 256];
    if (transpose_out) {
        int b = row / L1_LEN, l = row % L1_LEN;
        out[((size_t)(b * D_MODEL + tid)) * L1_LEN + l]       = v0;
        out[((size_t)(b * D_MODEL + tid + 256)) * L1_LEN + l] = v1;
    } else {
        out[(size_t)row * D_MODEL + tid]       = v0;
        out[(size_t)row * D_MODEL + tid + 256] = v1;
    }
}

// ---------------------------------------------------------------------------
extern "C" void kernel_launch(void* const* d_in, const int* in_sizes, int n_in,
                              void* d_out, int out_size, void* d_ws, size_t ws_size,
                              hipStream_t stream) {
    const float* x       = (const float*)d_in[0];
    const float* gf      = (const float*)d_in[1];
    const float* conv_w  = (const float*)d_in[2];
    const float* conv_b  = (const float*)d_in[3];
    const float* sWq = (const float*)d_in[4];  const float* sbq = (const float*)d_in[5];
    const float* sWk = (const float*)d_in[6];  const float* sbk = (const float*)d_in[7];
    const float* sWv = (const float*)d_in[8];  const float* sbv = (const float*)d_in[9];
    const float* sWo = (const float*)d_in[10]; const float* sbo = (const float*)d_in[11];
    const float* cWq = (const float*)d_in[12]; const float* cbq = (const float*)d_in[13];
    const float* cWk = (const float*)d_in[14]; const float* cbk = (const float*)d_in[15];
    const float* cWv = (const float*)d_in[16]; const float* cbv = (const float*)d_in[17];
    const float* cWo = (const float*)d_in[18]; const float* cbo = (const float*)d_in[19];
    const float* fw1 = (const float*)d_in[20]; const float* fb1 = (const float*)d_in[21];
    const float* fw2 = (const float*)d_in[22]; const float* fb2 = (const float*)d_in[23];
    const float* l1g = (const float*)d_in[24]; const float* l1b = (const float*)d_in[25];
    const float* l2g = (const float*)d_in[26]; const float* l2b = (const float*)d_in[27];
    const float* l3g = (const float*)d_in[28]; const float* l3b = (const float*)d_in[29];

    float* ws = (float*)d_ws;
    const size_t U = (size_t)1 << 20;            // 1M floats
    float* statT = ws;                           // [B*NW, 512]   1M
    float* cross = ws + 1 * U;                   // [B*NW, 512]   1M
    float* qb    = ws + 2 * U;                   // [B*L1, 512]   2M
    float* kb    = ws + 4 * U;                   // 2M
    float* vb    = ws + 6 * U;                   // 2M
    float* tmp   = ws + 8 * U;                   // raw attn out  2M
    float* proj  = ws + 10 * U;                  // attn proj     2M
    float* h1    = ws + 12 * U;                  // 2M
    float* h2    = ws + 14 * U;                  // 2M
    float* mid   = ws + 16 * U;                  // [B*L1, 2048]  8M
    float* fo    = ws + 24 * U;                  // 2M  (26M f32 total)

    dim3 blk(256);

    // 1) window stats (transposed) + pointwise conv as GEMM + relu
    winstat_kernel<<<4096, blk, 0, stream>>>(gf, statT);
    gemm_bias_act<<<dim3(2048 / 128, 512 / 64), blk, 0, stream>>>(statT, conv_w, conv_b, cross, 2048, 512, 512, 1);

    // 2) self-attention
    gemm_bias_act<<<dim3(4096 / 128, 512 / 64), blk, 0, stream>>>(x, sWq, sbq, qb, 4096, 512, 512, 0);
    gemm_bias_act<<<dim3(4096 / 128, 512 / 64), blk, 0, stream>>>(x, sWk, sbk, kb, 4096, 512, 512, 0);
    gemm_bias_act<<<dim3(4096 / 128, 512 / 64), blk, 0, stream>>>(x, sWv, sbv, vb, 4096, 512, 512, 0);
    flash_attn_kernel<<<512, blk, 0, stream>>>(qb, kb, vb, tmp, L1_LEN, L1_LEN);
    gemm_bias_act<<<dim3(4096 / 128, 512 / 64), blk, 0, stream>>>(tmp, sWo, sbo, proj, 4096, 512, 512, 0);
    add_ln_kernel<<<4096, blk, 0, stream>>>(x, proj, l1g, l1b, h1, 0);

    // 3) cross-attention (kv = 512 cross tokens)
    gemm_bias_act<<<dim3(4096 / 128, 512 / 64), blk, 0, stream>>>(h1, cWq, cbq, qb, 4096, 512, 512, 0);
    gemm_bias_act<<<dim3(2048 / 128, 512 / 64), blk, 0, stream>>>(cross, cWk, cbk, kb, 2048, 512, 512, 0);
    gemm_bias_act<<<dim3(2048 / 128, 512 / 64), blk, 0, stream>>>(cross, cWv, cbv, vb, 2048, 512, 512, 0);
    flash_attn_kernel<<<512, blk, 0, stream>>>(qb, kb, vb, tmp, L1_LEN, NW);
    gemm_bias_act<<<dim3(4096 / 128, 512 / 64), blk, 0, stream>>>(tmp, cWo, cbo, proj, 4096, 512, 512, 0);
    add_ln_kernel<<<4096, blk, 0, stream>>>(h1, proj, l2g, l2b, h2, 0);

    // 4) FFN + final LN (writes transposed [B, D, L1] directly to d_out)
    gemm_bias_act<<<dim3(4096 / 128, 2048 / 64), blk, 0, stream>>>(h2, fw1, fb1, mid, 4096, 2048, 512, 2);
    gemm_bias_act<<<dim3(4096 / 128, 512 / 64), blk, 0, stream>>>(mid, fw2, fb2, fo, 4096, 512, 2048, 0);
    add_ln_kernel<<<4096, blk, 0, stream>>>(h2, fo, l3g, l3b, (float*)d_out, 1);
}
